// CFConvAngular_2774548873970
// MI455X (gfx1250) — matrix operand order repaired and checked by reading, not compiled, not run on hardware
//
#include <hip/hip_runtime.h>
#include <hip/hip_bf16.h>

typedef __attribute__((ext_vector_type(16))) __bf16 v16bf;
typedef __attribute__((ext_vector_type(8)))  __bf16 v8bf;
typedef __attribute__((ext_vector_type(8)))  float  v8f;

#define B_   2
#define A_   512
#define T_   512
#define F_   128
#define RBF_ 25

#define LOG2E_ 1.4426950408889634f
#define LN2_   0.6931471805599453f

// shifted softplus: log(1+exp(v)) - log(2)
// = max(v,0) + ln2*log2(1 + 2^(-|v|*log2e)) - ln2
// log2 argument is in (1,2] -> raw v_log_f32 is safe (no denorm fixup needed),
// exp2 argument <= 0 -> raw v_exp_f32 is safe.
__device__ __forceinline__ float sspf(float v) {
    float t = __builtin_amdgcn_exp2f(-LOG2E_ * fabsf(v));   // e^{-|v|} in (0,1]
    float l = __builtin_amdgcn_logf(1.0f + t);              // log2(1+t) in (0,1]
    return fmaf(LN2_, l - 1.0f, fmaxf(v, 0.0f));
}

// ---------------------------------------------------------------------------
// Repack Wf1 (25x128, K zero-padded to 32) and Wf2 (128x128) into bf16
// B-fragment order for v_wmma_f32_16x16x32_bf16:
//   fragment = one 32(K) x 16(N) tile; 32 lanes x 16 bf16, stored so that
//   lane L reads its 16 values contiguously at frag_base + L*16.
//   Lanes 0-15: N = L, K = 0..15 ; lanes 16-31: N = L-16, K = 16..31.
// ---------------------------------------------------------------------------
__global__ void prep_weights(const float* __restrict__ Wf1,
                             const float* __restrict__ Wf2,
                             __bf16* __restrict__ wf1Frag,   // 8  frags * 512
                             __bf16* __restrict__ wf2Frag) { // 32 frags * 512
    int tid = blockIdx.x * blockDim.x + threadIdx.x;
    if (tid < 8 * 512) {
        int nt   = tid >> 9;
        int e    = tid & 511;
        int lane = e >> 4, i = e & 15;
        int half = lane >> 4;
        int n = nt * 16 + (lane & 15);
        int k = half * 16 + i;
        wf1Frag[tid] = (__bf16)((k < RBF_) ? Wf1[k * F_ + n] : 0.0f);
    } else if (tid < 8 * 512 + 32 * 512) {
        int t2   = tid - 8 * 512;
        int frag = t2 >> 9;              // kt*8 + nt
        int kt = frag >> 3, nt = frag & 7;
        int e    = t2 & 511;
        int lane = e >> 4, i = e & 15;
        int half = lane >> 4;
        int n = nt * 16 + (lane & 15);
        int k = kt * 32 + half * 16 + i;
        wf2Frag[t2] = (__bf16)Wf2[k * F_ + n];
    }
}

// ---------------------------------------------------------------------------
// y = x @ Win    (tiny [1024,128]x[128,128] GEMV, fp32)
// ---------------------------------------------------------------------------
__global__ void compute_y(const float* __restrict__ x,
                          const float* __restrict__ Win,
                          float* __restrict__ y) {
    __shared__ float xs[F_];
    int ba = blockIdx.x;
    int f  = threadIdx.x;
    xs[f] = x[ba * F_ + f];
    __syncthreads();
    float acc = 0.0f;
    #pragma unroll 4
    for (int i = 0; i < F_; ++i) acc += xs[i] * Win[i * F_ + f];
    y[ba * F_ + f] = acc;
}

// ---------------------------------------------------------------------------
// Fused: filter MLP (bf16 WMMA) + double gather + triple product + T-reduce
// + output GEMV + ssp.  One block per (b,a); 8 waves, 64 triplet rows each.
// nt loops kept rolled to bound VGPR pressure (<256, no spills / vgpr_msb).
// ---------------------------------------------------------------------------
__global__ __launch_bounds__(256)
void cfconv_main(const float* __restrict__ rij,
                 const float* __restrict__ mask,
                 const float* __restrict__ bf1,
                 const float* __restrict__ bf2,
                 const float* __restrict__ Wout,
                 const float* __restrict__ bout,
                 const int*   __restrict__ nbrJ,
                 const int*   __restrict__ nbrK,
                 const float* __restrict__ y,
                 const __bf16* __restrict__ wf1Frag,
                 const __bf16* __restrict__ wf2Frag,
                 float* __restrict__ out) {
    __shared__ float sOut[F_];
    __shared__ __align__(16) __bf16 sStage[8][16 * F_];   // 4KB per wave

    const int ba   = blockIdx.x;
    const int b    = ba / A_;
    const int tid  = threadIdx.x;
    const int w    = tid >> 5;         // wave id (wave32)
    const int lane = tid & 31;
    const int half = lane >> 4;
    const int m    = lane & 15;

    if (tid < F_) sOut[tid] = 0.0f;
    __syncthreads();

    float accF[8];
    #pragma unroll
    for (int i = 0; i < 8; ++i) accF[i] = 0.0f;

    #pragma unroll 1
    for (int tt = 0; tt < 4; ++tt) {
        const int tBase = w * 64 + tt * 16;

        // ---- triplet metadata for this 16-row tile (shared by all nt)
        int   jIdx[8], kIdx[8];
        float mIdx[8];
        #pragma unroll
        for (int r = 0; r < 8; ++r) {
            int p = ba * T_ + tBase + r + 8 * half;
            jIdx[r] = nbrJ[p];
            kIdx[r] = nbrK[p];
            mIdx[r] = mask[p];
        }

        // ---- GEMM1 A-fragment straight from r_ij (fp32 -> bf16, K pad to 32)
        // A 16x32 in contiguous k order (same as wf1Frag): lane<16 holds K {0..15}, lane>=16 holds
        // K {16..31}; row M = lane%16.
        v16bf a1;
        {
            const float* rrow = rij + ((size_t)ba * T_ + (tBase + m)) * RBF_;
            #pragma unroll
            for (int i = 0; i < 16; ++i) {
                int k = half * 16 + i;
                a1[i] = (k < RBF_) ? (__bf16)rrow[k] : (__bf16)0.0f;
            }
        }

        // ---- GEMM1 (K=32 in one WMMA) + bias + ssp -> bf16 stage in LDS
        #pragma unroll 1
        for (int nt = 0; nt < 8; ++nt) {
            v16bf bfr = *(const v16bf*)(wf1Frag + nt * 512 + lane * 16);
            v8f c = {};
            c = __builtin_amdgcn_wmma_f32_16x16x32_bf16(false, a1, false, bfr,
                                                        (short)0, c, false, false);
            float b1 = bf1[nt * 16 + m];
            // C layout: lanes 0-15 -> N=lane, M=r ; lanes 16-31 -> N=lane-16, M=8+r
            #pragma unroll
            for (int r = 0; r < 8; ++r) {
                float v = sspf(c[r] + b1);
                sStage[w][(r + 8 * half) * F_ + nt * 16 + m] = (__bf16)v;
            }
        }
        // same-wave DS ops are in-order: stage stores complete before reloads

        // ---- GEMM2 A-fragments from LDS stage (two aligned b128 per kt)
        v16bf a2[4];
        #pragma unroll
        for (int kt = 0; kt < 4; ++kt) {
            const __bf16* rowp = &sStage[w][m * F_ + kt * 32 + half * 16];
            v8bf c0 = *(const v8bf*)(rowp);
            v8bf c1 = *(const v8bf*)(rowp + 8);
            #pragma unroll
            for (int i = 0; i < 8; ++i) { a2[kt][i] = c0[i]; a2[kt][i + 8] = c1[i]; }
        }

        // ---- GEMM2 (K=128 -> 4 WMMAs) fused with gather*product*reduce
        #pragma unroll 1
        for (int nt = 0; nt < 8; ++nt) {
            v8f c = {};
            #pragma unroll
            for (int kt = 0; kt < 4; ++kt) {
                v16bf bfr = *(const v16bf*)(wf2Frag + (kt * 8 + nt) * 512 + lane * 16);
                c = __builtin_amdgcn_wmma_f32_16x16x32_bf16(false, a2[kt], false, bfr,
                                                            (short)0, c, false, false);
            }
            const int   f  = nt * 16 + m;
            const float b2 = bf2[f];
            float a = 0.0f;
            #pragma unroll
            for (int r = 0; r < 8; ++r) {
                float yj = y[(b * A_ + jIdx[r]) * F_ + f];   // L2-resident table
                float yk = y[(b * A_ + kIdx[r]) * F_ + f];
                a += (c[r] + b2) * yj * yk * mIdx[r];
            }
            accF[nt] += a;
        }
    }

    // per-feature reduce across waves / lane halves (ds_add_f32)
    #pragma unroll
    for (int nt = 0; nt < 8; ++nt)
        atomicAdd(&sOut[nt * 16 + m], accF[nt]);
    __syncthreads();

    // ---- out = ssp(yAcc @ Wout + bout)
    if (tid < F_) {
        const int f = tid;
        float acc = bout[f];
        #pragma unroll 4
        for (int g = 0; g < F_; ++g) acc += sOut[g] * Wout[g * F_ + f];
        out[ba * F_ + f] = sspf(acc);
    }
}

// ---------------------------------------------------------------------------
extern "C" void kernel_launch(void* const* d_in, const int* in_sizes, int n_in,
                              void* d_out, int out_size, void* d_ws, size_t ws_size,
                              hipStream_t stream) {
    const float* x    = (const float*)d_in[0];
    const float* rij  = (const float*)d_in[1];
    const float* mask = (const float*)d_in[2];
    const float* Wf1  = (const float*)d_in[3];
    const float* bf1  = (const float*)d_in[4];
    const float* Wf2  = (const float*)d_in[5];
    const float* bf2  = (const float*)d_in[6];
    const float* Win  = (const float*)d_in[7];
    const float* Wout = (const float*)d_in[8];
    const float* bout = (const float*)d_in[9];
    const int*   nbrJ = (const int*)d_in[10];
    const int*   nbrK = (const int*)d_in[11];
    float* out = (float*)d_out;

    // workspace layout (~552 KB): y table, then bf16 weight fragments
    char*   ws      = (char*)d_ws;
    float*  y       = (float*)ws;                                   // 512 KB
    __bf16* wf1Frag = (__bf16*)(ws + (size_t)B_ * A_ * F_ * sizeof(float));
    __bf16* wf2Frag = wf1Frag + 8 * 512;                            // +8KB, 32KB

    prep_weights<<<80, 256, 0, stream>>>(Wf1, Wf2, wf1Frag, wf2Frag);
    compute_y<<<B_ * A_, F_, 0, stream>>>(x, Win, y);
    cfconv_main<<<B_ * A_, 256, 0, stream>>>(rij, mask, bf1, bf2, Wout, bout,
                                             nbrJ, nbrK, y, wf1Frag, wf2Frag, out);
}
